// HierGRU_77721728188663
// MI455X (gfx1250) — compile-verified, matrix operand order checked by reading, not run on hardware
//
#include <hip/hip_runtime.h>

// ---------------- types ----------------
typedef __attribute__((ext_vector_type(16))) __bf16 bf16x16;
typedef __attribute__((ext_vector_type(8)))  __bf16 bf16x8;
typedef __attribute__((ext_vector_type(8)))  float  f32x8;

union BF16x16 { bf16x16 v; bf16x8 h[2]; };

__device__ __forceinline__ f32x8 wmma_bf16(bf16x16 a, bf16x16 b, f32x8 c) {
  // (neg_a, A, neg_b, B, c_mod, C, reuse_a, reuse_b)
  return __builtin_amdgcn_wmma_f32_16x16x32_bf16(false, a, false, b, (short)0, c,
                                                 false, false);
}

__device__ __forceinline__ float sigmoidf_(float x) {
  return 1.0f / (1.0f + __expf(-x));
}

// ---------------- problem constants ----------------
constexpr int BSZ  = 32;
constexpr int T    = 2048;
constexpr int L    = 32;          // win_len
constexpr int NW   = 64;          // windows per batch
constexpr int NWIN = BSZ * NW;    // 2048 windows
constexpr int ROWS = BSZ * T;     // 65536 (= NWIN * L)
constexpr int TOKD = 128, ARGD = 16;
constexpr int DIN  = 146, DINP = 160;          // padded to mult of 32
constexpr int WH   = 128, G3  = 3 * WH;        // 384
constexpr int SH   = 192, SG3 = 3 * SH;        // 576
constexpr int SDIN = 2 * WH;                   // 256

// ---------------- weight pack: f32 -> bf16 with K padding ----------------
__global__ void pack_bf16(const float* __restrict__ src, __bf16* __restrict__ dst,
                          int rows, int cols, int colsPad) {
  int idx = blockIdx.x * blockDim.x + threadIdx.x;
  if (idx >= rows * colsPad) return;
  int r = idx / colsPad, c = idx - r * colsPad;
  float v = (c < cols) ? src[(long)r * cols + c] : 0.0f;
  dst[idx] = (__bf16)v;
}

// ---------------- embed + concat -> bf16 X [ROWS x DINP] ----------------
__global__ void embed_concat(const int* __restrict__ tok_ids,
                             const int* __restrict__ arg_ids,
                             const float* __restrict__ burst,
                             const float* __restrict__ pos,
                             const float* __restrict__ tok_emb,
                             const float* __restrict__ arg_emb,
                             __bf16* __restrict__ xbf) {
  long idx = (long)blockIdx.x * blockDim.x + threadIdx.x;
  if (idx >= (long)ROWS * DINP) return;
  int r = (int)(idx / DINP), c = (int)(idx - (long)r * DINP);
  float v;
  if (c < TOKD)             v = tok_emb[(long)tok_ids[r] * TOKD + c];
  else if (c < TOKD + ARGD) v = arg_emb[(long)arg_ids[r] * ARGD + (c - TOKD)];
  else if (c == 144)        v = burst[r];
  else if (c == 145)        v = pos[r];
  else                      v = 0.0f;
  xbf[idx] = (__bf16)v;
}

// ---------------- wave-per-supertile WMMA GEMM: C = A * Bw^T + bias --------
// Register blocking: each wave computes 2 row-tiles x 3 col-tiles (the three
// GRU gates), i.e. 6 WMMAs per 5 tile-loads per K-step.
// A: [M x lda] bf16 row-major; Bw: [N x ldb] bf16 row-major; C: [M x ldc] f32.
// Requirements: M % 32 == 0, (N/16) % 3 == 0, K % 32 == 0.
__global__ void __launch_bounds__(256)
gemm_bias(const __bf16* __restrict__ A, const __bf16* __restrict__ Bw,
          const float* __restrict__ bias, float* __restrict__ C,
          int M, int N, int K, int lda, int ldb, int ldc) {
  int wid  = (int)((blockIdx.x * blockDim.x + threadIdx.x) >> 5);
  int lane = threadIdx.x & 31;
  int nct3 = (N >> 4) / 3;                 // col-tile groups per gate
  int rowp = wid / nct3, cg = wid - rowp * nct3;
  if (rowp >= (M >> 5)) return;
  int l = lane & 15, half = lane >> 4;

  const __bf16* arow0 = A + (long)(rowp * 32 + l) * lda;
  const __bf16* arow1 = arow0 + (long)16 * lda;
  const __bf16* brow[3];
#pragma unroll
  for (int j = 0; j < 3; ++j)
    brow[j] = Bw + (long)((cg + j * nct3) * 16 + l) * ldb;

  f32x8 acc[2][3] = {};
  int nkt = K >> 5;
  for (int kt = 0; kt < nkt; ++kt) {
    int k0 = kt * 32 + half * 8;
    BF16x16 a0, a1, b[3];
    a0.h[0] = *(const bf16x8*)(arow0 + k0);
    a0.h[1] = *(const bf16x8*)(arow0 + k0 + 16);
    a1.h[0] = *(const bf16x8*)(arow1 + k0);
    a1.h[1] = *(const bf16x8*)(arow1 + k0 + 16);
#pragma unroll
    for (int j = 0; j < 3; ++j) {
      b[j].h[0] = *(const bf16x8*)(brow[j] + k0);
      b[j].h[1] = *(const bf16x8*)(brow[j] + k0 + 16);
    }
#pragma unroll
    for (int j = 0; j < 3; ++j) {
      acc[0][j] = wmma_bf16(a0.v, b[j].v, acc[0][j]);
      acc[1][j] = wmma_bf16(a1.v, b[j].v, acc[1][j]);
    }
  }
#pragma unroll
  for (int j = 0; j < 3; ++j) {
    int col = (cg + j * nct3) * 16 + l;
    float bv = bias[col];
#pragma unroll
    for (int rb = 0; rb < 2; ++rb) {
      float* cc = C + (long)(rowp * 32 + rb * 16 + half * 8) * ldc + col;
#pragma unroll
      for (int i = 0; i < 8; ++i) cc[(long)i * ldc] = acc[rb][j][i] + bv;
    }
  }
}

// ---------------- window GRU recurrence (one direction) ----------------
// 16 windows per workgroup; 8 waves; wave w owns gate-column slice
// [w*16,(w+1)*16) of r/z/n.  Whh (96 KB bf16) is preloaded into LDS and
// re-fed to WMMA via ds_load_b128 for all 32 timesteps; h lives in LDS too.
__global__ void __launch_bounds__(256)
win_gru(const float* __restrict__ gx, const __bf16* __restrict__ whhp,
        const float* __restrict__ bhh, __bf16* __restrict__ win, int reverse) {
  extern __shared__ __bf16 wsh[];          // [G3 * WH] = 96 KB
  __shared__ __bf16 hbf[16][WH];           // 4 KB
  __shared__ float  hf [16][WH];           // 8 KB
  int tid = threadIdx.x;

  // preload Whh (bf16) into LDS, vectorized 16B copies
  {
    const uint4* src = (const uint4*)whhp;
    uint4* dst = (uint4*)wsh;
    for (int i = tid; i < (G3 * WH) / 8; i += 256) dst[i] = src[i];
  }
  for (int i = tid; i < 16 * WH; i += 256) {
    ((__bf16*)hbf)[i] = (__bf16)0.0f;
    ((float*)hf)[i]   = 0.0f;
  }
  __syncthreads();

  int w = tid >> 5, lane = tid & 31, l = lane & 15, half = lane >> 4;
  int colg = w * 16 + l;                 // column within one gate [0,128)
  long g0 = (long)blockIdx.x * 16;       // first window of this tile

  const float bhr = bhh[colg], bhz = bhh[WH + colg], bhn = bhh[2 * WH + colg];
  const __bf16* br = wsh + (long)(colg) * WH;            // r rows (LDS)
  const __bf16* bz = wsh + (long)(WH  + colg) * WH;      // z rows
  const __bf16* bn = wsh + (long)(2*WH + colg) * WH;     // n rows

  float pool[8] = {0, 0, 0, 0, 0, 0, 0, 0};

  for (int t = 0; t < L; ++t) {
    f32x8 ar = {}, az = {}, an = {};
#pragma unroll
    for (int kt = 0; kt < WH / 32; ++kt) {
      int k0 = kt * 32 + half * 8;
      BF16x16 a, b0, b1, b2;
      a.h[0]  = *(const bf16x8*)&hbf[l][k0];
      a.h[1]  = *(const bf16x8*)&hbf[l][k0 + 16];
      b0.h[0] = *(const bf16x8*)(br + k0); b0.h[1] = *(const bf16x8*)(br + k0 + 16);
      b1.h[0] = *(const bf16x8*)(bz + k0); b1.h[1] = *(const bf16x8*)(bz + k0 + 16);
      b2.h[0] = *(const bf16x8*)(bn + k0); b2.h[1] = *(const bf16x8*)(bn + k0 + 16);
      ar = wmma_bf16(a.v, b0.v, ar);
      az = wmma_bf16(a.v, b1.v, az);
      an = wmma_bf16(a.v, b2.v, an);
    }
    int tt = reverse ? (L - 1 - t) : t;
    float hn_[8];
#pragma unroll
    for (int i = 0; i < 8; ++i) {
      int m = half * 8 + i;                        // window within tile
      long grow = (g0 + m) * L + tt;               // row in gx
      const float* gr = gx + grow * G3;
      float r = sigmoidf_(gr[colg]          + ar[i] + bhr);
      float z = sigmoidf_(gr[WH + colg]     + az[i] + bhz);
      float n = tanhf(gr[2 * WH + colg] + r * (an[i] + bhn));
      float h = (1.0f - z) * n + z * hf[m][colg];
      hn_[i] = h;
      pool[i] += h;
    }
    __syncthreads();   // all reads of h done before overwrite
#pragma unroll
    for (int i = 0; i < 8; ++i) {
      int m = half * 8 + i;
      hf[m][colg]  = hn_[i];
      hbf[m][colg] = (__bf16)hn_[i];
    }
    __syncthreads();   // writes visible before next step's WMMA reads
  }
  int outc = reverse * WH + colg;   // concat [out_f, out_b]
#pragma unroll
  for (int i = 0; i < 8; ++i) {
    int m = half * 8 + i;
    win[(g0 + m) * SDIN + outc] = (__bf16)(pool[i] * (1.0f / L));
  }
}

// ---------------- sequence GRU recurrence ----------------
// 16 batch rows per workgroup (2 workgroups); 12 waves; wave w owns gate
// slice [w*16,(w+1)*16).  s_whh (221 KB bf16) preloaded into LDS (one WG
// per WGP -- only 2 workgroups exist).
__global__ void __launch_bounds__(384)
seq_gru(const float* __restrict__ gxs, const __bf16* __restrict__ swhhp,
        const float* __restrict__ bhh, float* __restrict__ out) {
  extern __shared__ __bf16 swsh[];         // [SG3 * SH] = 221 KB
  __shared__ __bf16 hbf[16][SH];           // 6 KB
  __shared__ float  hf [16][SH];           // 12 KB
  int tid = threadIdx.x;

  {
    const uint4* src = (const uint4*)swhhp;
    uint4* dst = (uint4*)swsh;
    for (int i = tid; i < (SG3 * SH) / 8; i += 384) dst[i] = src[i];
  }
  for (int i = tid; i < 16 * SH; i += 384) {
    ((__bf16*)hbf)[i] = (__bf16)0.0f;
    ((float*)hf)[i]   = 0.0f;
  }
  __syncthreads();

  int w = tid >> 5, lane = tid & 31, l = lane & 15, half = lane >> 4;
  int colg = w * 16 + l;                // [0,192)
  long b0 = (long)blockIdx.x * 16;

  const float bhr = bhh[colg], bhz = bhh[SH + colg], bhn = bhh[2 * SH + colg];
  const __bf16* br = swsh + (long)(colg) * SH;
  const __bf16* bz = swsh + (long)(SH  + colg) * SH;
  const __bf16* bn = swsh + (long)(2*SH + colg) * SH;

  for (int t = 0; t < NW; ++t) {
    f32x8 ar = {}, az = {}, an = {};
#pragma unroll
    for (int kt = 0; kt < SH / 32; ++kt) {
      int k0 = kt * 32 + half * 8;
      BF16x16 a, b0v, b1v, b2v;
      a.h[0]   = *(const bf16x8*)&hbf[l][k0];
      a.h[1]   = *(const bf16x8*)&hbf[l][k0 + 16];
      b0v.h[0] = *(const bf16x8*)(br + k0); b0v.h[1] = *(const bf16x8*)(br + k0 + 16);
      b1v.h[0] = *(const bf16x8*)(bz + k0); b1v.h[1] = *(const bf16x8*)(bz + k0 + 16);
      b2v.h[0] = *(const bf16x8*)(bn + k0); b2v.h[1] = *(const bf16x8*)(bn + k0 + 16);
      ar = wmma_bf16(a.v, b0v.v, ar);
      az = wmma_bf16(a.v, b1v.v, az);
      an = wmma_bf16(a.v, b2v.v, an);
    }
    float hn_[8];
#pragma unroll
    for (int i = 0; i < 8; ++i) {
      int m = half * 8 + i;
      long grow = (b0 + m) * NW + t;
      const float* gr = gxs + grow * SG3;
      float r = sigmoidf_(gr[colg]          + ar[i] + bhr);
      float z = sigmoidf_(gr[SH + colg]     + az[i] + bhz);
      float n = tanhf(gr[2 * SH + colg] + r * (an[i] + bhn));
      hn_[i] = (1.0f - z) * n + z * hf[m][colg];
    }
    __syncthreads();
#pragma unroll
    for (int i = 0; i < 8; ++i) {
      int m = half * 8 + i;
      hf[m][colg]  = hn_[i];
      hbf[m][colg] = (__bf16)hn_[i];
    }
    __syncthreads();
  }
#pragma unroll
  for (int i = 0; i < 8; ++i) {
    int m = half * 8 + i;
    out[(b0 + m) * SH + colg] = hf[m][colg];
  }
}

// ---------------- host launcher ----------------
extern "C" void kernel_launch(void* const* d_in, const int* in_sizes, int n_in,
                              void* d_out, int out_size, void* d_ws, size_t ws_size,
                              hipStream_t stream) {
  const int*   tok_ids = (const int*)d_in[0];
  const int*   arg_ids = (const int*)d_in[1];
  const float* burst   = (const float*)d_in[2];
  const float* pos     = (const float*)d_in[3];
  const float* tok_emb = (const float*)d_in[4];
  const float* arg_emb = (const float*)d_in[5];
  const float* wih_f   = (const float*)d_in[6];
  const float* whh_f   = (const float*)d_in[7];
  const float* bih_f   = (const float*)d_in[8];
  const float* bhh_f   = (const float*)d_in[9];
  const float* wih_b   = (const float*)d_in[10];
  const float* whh_b   = (const float*)d_in[11];
  const float* bih_b   = (const float*)d_in[12];
  const float* bhh_b   = (const float*)d_in[13];
  const float* s_wih   = (const float*)d_in[14];
  const float* s_whh   = (const float*)d_in[15];
  const float* s_bih   = (const float*)d_in[16];
  const float* s_bhh   = (const float*)d_in[17];
  float* out = (float*)d_out;

  char* ws = (char*)d_ws;
  size_t off = 0;
  auto carve = [&](size_t bytes) -> void* {
    void* p = ws + off;
    off += (bytes + 255) & ~(size_t)255;
    return p;
  };
  __bf16* xbf    = (__bf16*)carve((size_t)ROWS * DINP * 2);   // ~21 MB
  float*  gx     = (float*) carve((size_t)ROWS * G3 * 4);     // ~100 MB (reused per dir)
  __bf16* winbf  = (__bf16*)carve((size_t)NWIN * SDIN * 2);   // 1 MB
  float*  gxs    = (float*) carve((size_t)NWIN * SG3 * 4);    // ~4.7 MB
  __bf16* wihp_f = (__bf16*)carve((size_t)G3 * DINP * 2);
  __bf16* wihp_b = (__bf16*)carve((size_t)G3 * DINP * 2);
  __bf16* whhp_f = (__bf16*)carve((size_t)G3 * WH * 2);
  __bf16* whhp_b = (__bf16*)carve((size_t)G3 * WH * 2);
  __bf16* swihp  = (__bf16*)carve((size_t)SG3 * SDIN * 2);
  __bf16* swhhp  = (__bf16*)carve((size_t)SG3 * SH * 2);

  auto cdiv = [](long a, long b) { return (int)((a + b - 1) / b); };

  // 1) weight packing (f32 -> bf16, pad K)
  pack_bf16<<<cdiv((long)G3 * DINP, 256), 256, 0, stream>>>(wih_f, wihp_f, G3, DIN, DINP);
  pack_bf16<<<cdiv((long)G3 * DINP, 256), 256, 0, stream>>>(wih_b, wihp_b, G3, DIN, DINP);
  pack_bf16<<<cdiv((long)G3 * WH,   256), 256, 0, stream>>>(whh_f, whhp_f, G3, WH, WH);
  pack_bf16<<<cdiv((long)G3 * WH,   256), 256, 0, stream>>>(whh_b, whhp_b, G3, WH, WH);
  pack_bf16<<<cdiv((long)SG3 * SDIN,256), 256, 0, stream>>>(s_wih, swihp, SG3, SDIN, SDIN);
  pack_bf16<<<cdiv((long)SG3 * SH,  256), 256, 0, stream>>>(s_whh, swhhp, SG3, SH, SH);

  // 2) embed + concat
  embed_concat<<<cdiv((long)ROWS * DINP, 256), 256, 0, stream>>>(
      tok_ids, arg_ids, burst, pos, tok_emb, arg_emb, xbf);

  // 3) forward window direction: gx GEMM + recurrence
  {
    long tiles = (long)(ROWS / 32) * ((G3 / 16) / 3);
    gemm_bias<<<cdiv(tiles * 32, 256), 256, 0, stream>>>(
        xbf, wihp_f, bih_f, gx, ROWS, G3, DINP, DINP, DINP, G3);
    win_gru<<<NWIN / 16, 256, (size_t)G3 * WH * 2, stream>>>(
        gx, whhp_f, bhh_f, winbf, 0);
  }
  // 4) backward window direction (reuses gx buffer; stream-ordered)
  {
    long tiles = (long)(ROWS / 32) * ((G3 / 16) / 3);
    gemm_bias<<<cdiv(tiles * 32, 256), 256, 0, stream>>>(
        xbf, wihp_b, bih_b, gx, ROWS, G3, DINP, DINP, DINP, G3);
    win_gru<<<NWIN / 16, 256, (size_t)G3 * WH * 2, stream>>>(
        gx, whhp_b, bhh_b, winbf, 1);
  }

  // 5) sequence GRU: gx GEMM + recurrence -> final hidden to d_out
  {
    long tiles = (long)(NWIN / 32) * ((SG3 / 16) / 3);
    gemm_bias<<<cdiv(tiles * 32, 256), 256, 0, stream>>>(
        winbf, swihp, s_bih, gxs, NWIN, SG3, SDIN, SDIN, SDIN, SG3);
    seq_gru<<<BSZ / 16, 384, (size_t)SG3 * SH * 2, stream>>>(
        gxs, swhhp, s_bhh, out);
  }
}
